// AttrDecoder_19404662243661
// MI455X (gfx1250) — matrix operand order repaired and checked by reading, not compiled, not run on hardware
//
#include <hip/hip_runtime.h>
#include <math.h>

// Problem constants (from the reference)
#define A_ATTR 312
#define LAT    64
#define HID    128
#define BATCH  8192

#define MTILE   256   // batch rows per workgroup (8 waves x 32 rows each)
#define THREADS 256
#define PITCH   72    // bf16 elems per LDS row: 144 B (16B aligned, 36-bank stride)

typedef __attribute__((ext_vector_type(16))) __bf16        v16bf;
typedef __attribute__((ext_vector_type(8)))  float         v8f;
typedef __attribute__((ext_vector_type(8)))  __bf16        bf16x8;
typedef __attribute__((ext_vector_type(4)))  unsigned int  uint4v;

__global__ __launch_bounds__(THREADS)
void attr_decoder_kernel(const float* __restrict__ x,   // [B, A*LAT]
                         const float* __restrict__ W1,  // [A, LAT, HID]
                         const float* __restrict__ b1,  // [A, HID]
                         const float* __restrict__ W2,  // [A, HID, 1]
                         const float* __restrict__ b2,  // [A, 1]
                         float* __restrict__ out)       // [B, A]
{
    const int a   = blockIdx.y;          // attribute
    const int b0  = blockIdx.x * MTILE;  // batch tile base row
    const int tid = threadIdx.x;

    __shared__ __bf16 xs[MTILE * PITCH]; // x tile   [m][k]   (36.9 KB)
    __shared__ __bf16 ws[HID   * PITCH]; // W1^T     [n][k]   (18.4 KB)
    __shared__ float  b1s[HID];
    __shared__ float  w2s[HID];

    // ---- stage x tile: 256 rows x 64 cols, f32 -> bf16, 16B LDS stores ----
    #pragma unroll
    for (int i = 0; i < 8; ++i) {
        int e  = tid + i * THREADS;   // 0..2047 tasks of 8 floats
        int m  = e >> 3;
        int k8 = e & 7;
        const float4* src = reinterpret_cast<const float4*>(
            x + (size_t)(b0 + m) * (A_ATTR * LAT) + (size_t)a * LAT + k8 * 8);
        float4 v0 = src[0];
        float4 v1 = src[1];
        bf16x8 w;
        w[0] = (__bf16)v0.x; w[1] = (__bf16)v0.y; w[2] = (__bf16)v0.z; w[3] = (__bf16)v0.w;
        w[4] = (__bf16)v1.x; w[5] = (__bf16)v1.y; w[6] = (__bf16)v1.z; w[7] = (__bf16)v1.w;
        *reinterpret_cast<bf16x8*>(&xs[m * PITCH + k8 * 8]) = w;
    }

    // ---- stage W1_a transposed to [n=HID][k=LAT]; lane-coalesced b32 reads,
    //      one aligned 16B LDS store per task ----
    {
        const float* wg = W1 + (size_t)a * LAT * HID;
        #pragma unroll
        for (int i = 0; i < 4; ++i) {
            int e  = tid + i * THREADS;   // 0..1023 tasks of 8 K-values
            int l8 = e >> 7;              // K-octet 0..7
            int h  = e & 127;             // consecutive lanes -> consecutive h (coalesced)
            bf16x8 w;
            #pragma unroll
            for (int j = 0; j < 8; ++j)
                w[j] = (__bf16)wg[(l8 * 8 + j) * HID + h];
            *reinterpret_cast<bf16x8*>(&ws[h * PITCH + l8 * 8]) = w;
        }
    }
    if (tid < HID) {
        b1s[tid] = b1[(size_t)a * HID + tid];
        w2s[tid] = W2[(size_t)a * HID + tid];   // OUT==1 -> [A,HID]
    }
    __syncthreads();

    // ---------------- per-wave 32x128 strip via WMMA ------------------------
    const int wave  = tid >> 5;
    const int lane  = tid & 31;
    const int lhalf = lane >> 4;   // 0: lanes 0-15, 1: lanes 16-31
    const int l16   = lane & 15;
    const int m0    = wave * 32;   // two 16-row M-tiles per wave

    // A frag (16-bit 16x32): lane row = l16; lanes 0-15 take K {0..7,16..23},
    // lanes 16-31 take K {8..15,24..31} (handled by lhalf*8 in the base).
    const __bf16* pA = &xs[(m0 + l16) * PITCH + lhalf * 8];
    // B frag (16-bit 32x16): lane col = l16; lanes 0-15: K {0..7,16..23}, lanes 16-31: K {8..15,24..31}.
    const __bf16* pB = &ws[l16 * PITCH + lhalf * 8];

    v8f c[2][8];
    #pragma unroll
    for (int mt = 0; mt < 2; ++mt)
        #pragma unroll
        for (int t = 0; t < 8; ++t)
            c[mt][t] = (v8f){0.f,0.f,0.f,0.f,0.f,0.f,0.f,0.f};

    #pragma unroll
    for (int kc = 0; kc < 2; ++kc) {              // LAT = 2 K-chunks of 32
        union { v16bf v; uint4v q[2]; } a0, a1;
        a0.q[0] = *reinterpret_cast<const uint4v*>(pA + kc * 32);
        a0.q[1] = *reinterpret_cast<const uint4v*>(pA + kc * 32 + 16);
        a1.q[0] = *reinterpret_cast<const uint4v*>(pA + 16 * PITCH + kc * 32);
        a1.q[1] = *reinterpret_cast<const uint4v*>(pA + 16 * PITCH + kc * 32 + 16);

        #pragma unroll
        for (int t = 0; t < 8; ++t) {             // HID = 8 N-tiles of 16
            union { v16bf v; uint4v q[2]; } bfrag;  // 8 bf16 at +0 and 8 at +16 = 32 B
            bfrag.q[0] = *reinterpret_cast<const uint4v*>(pB + t * 16 * PITCH + kc * 32);
            bfrag.q[1] = *reinterpret_cast<const uint4v*>(pB + t * 16 * PITCH + kc * 32 + 16);

            c[0][t] = __builtin_amdgcn_wmma_f32_16x16x32_bf16(
                          false, a0.v, false, bfrag.v, (short)0, c[0][t], false, false);
            c[1][t] = __builtin_amdgcn_wmma_f32_16x16x32_bf16(
                          false, a1.v, false, bfrag.v, (short)0, c[1][t], false, false);
        }
    }

    // ---------------- epilogue: bias + relu + dot(w2) + sigmoid -------------
    // C/D layout: VGPR r, lanes 0-15 -> (M=r, N=lane), lanes 16-31 -> (M=r+8, N=lane-16)
    const float bb2 = b2[a];
    #pragma unroll
    for (int mt = 0; mt < 2; ++mt) {
        float s[8];
        #pragma unroll
        for (int r = 0; r < 8; ++r) s[r] = 0.f;

        #pragma unroll
        for (int t = 0; t < 8; ++t) {
            const int   n    = t * 16 + l16;   // this lane's column in tile t
            const float bias = b1s[n];
            const float w    = w2s[n];
            #pragma unroll
            for (int r = 0; r < 8; ++r) {
                float h = c[mt][t][r] + bias;
                h = h > 0.f ? h : 0.f;          // relu
                s[r] += h * w;
            }
        }
        // reduce over N within each 16-lane half (halves hold disjoint rows)
        #pragma unroll
        for (int r = 0; r < 8; ++r) {
            float v = s[r];
            v += __shfl_xor(v, 1, 32);
            v += __shfl_xor(v, 2, 32);
            v += __shfl_xor(v, 4, 32);
            v += __shfl_xor(v, 8, 32);
            s[r] = v;
        }
        const int rowbase = b0 + m0 + mt * 16 + lhalf * 8;
        #pragma unroll
        for (int r = 0; r < 8; ++r) {
            if (l16 == r) {
                float z = s[r] + bb2;
                out[(size_t)(rowbase + r) * A_ATTR + a] = 1.0f / (1.0f + __expf(-z));
            }
        }
    }
}

extern "C" void kernel_launch(void* const* d_in, const int* in_sizes, int n_in,
                              void* d_out, int out_size, void* d_ws, size_t ws_size,
                              hipStream_t stream) {
    const float* x  = (const float*)d_in[0];
    const float* W1 = (const float*)d_in[1];
    const float* b1 = (const float*)d_in[2];
    const float* W2 = (const float*)d_in[3];
    const float* b2 = (const float*)d_in[4];
    float* out = (float*)d_out;

    dim3 grid(BATCH / MTILE, A_ATTR);   // 32 x 312 workgroups
    attr_decoder_kernel<<<grid, THREADS, 0, stream>>>(x, W1, b1, W2, b2, out);
}
